// SimCLR_Loss_73340861546691
// MI455X (gfx1250) — matrix operand order repaired and checked by reading, not compiled, not run on hardware
//
#include <hip/hip_runtime.h>
#include <math.h>

// ---------------------------------------------------------------------------
// SimCLR NT-Xent loss, B=16, D=128, M=2048, N=2M=4096, TEMP=0.5
// Compute-bound on the N x N Gram matrix -> drive it through
// v_wmma_f32_16x16x32_bf16 with bf16 hi/lo fp32-emulation (3 products).
// ---------------------------------------------------------------------------

#define BATCH 16
#define DDIM  128
#define MCOLS 2048
#define NROWS 4096            // 2 * MCOLS

typedef __attribute__((ext_vector_type(16))) __bf16       v16bf;
typedef __attribute__((ext_vector_type(8)))  float        v8f;
typedef __attribute__((ext_vector_type(8)))  unsigned int u32x8;   // 32 bytes
typedef __attribute__((ext_vector_type(4)))  unsigned int u32x4;   // 16 bytes

union FragB { u32x8 u;    v16bf b; };   // one 32B load  -> 16 bf16
union FragA { u32x4 h[2]; v16bf b; };   // two 16B loads -> 16 bf16

static __device__ __forceinline__ unsigned short f32_to_bf16_rne(float f) {
  unsigned int u = __float_as_uint(f);
  unsigned int r = u + 0x7FFFu + ((u >> 16) & 1u);   // round-to-nearest-even
  return (unsigned short)(r >> 16);
}
static __device__ __forceinline__ float bf16_to_f32(unsigned short h) {
  return __uint_as_float(((unsigned int)h) << 16);
}

// ---------------------------------------------------------------------------
// Phase 1: L2-normalize every row (fp32), emit bf16 hi/lo split + positive
// logits. zn layout: [B][N][D] row-major, row m = z_i column m, row M+m = z_j.
// ---------------------------------------------------------------------------
__global__ void simclr_normalize(const float* __restrict__ Zi,
                                 const float* __restrict__ Zj,
                                 unsigned short* __restrict__ znHi,
                                 unsigned short* __restrict__ znLo,
                                 float* __restrict__ pos) {
  int gid = blockIdx.x * blockDim.x + threadIdx.x;
  if (gid >= BATCH * MCOLS) return;
  int b = gid / MCOLS, m = gid % MCOLS;
  const float* zi = Zi + (size_t)b * DDIM * MCOLS + m;   // stride MCOLS over d
  const float* zj = Zj + (size_t)b * DDIM * MCOLS + m;

  float ssi = 0.f, ssj = 0.f;
  for (int d = 0; d < DDIM; ++d) {
    float xi = zi[(size_t)d * MCOLS]; ssi = fmaf(xi, xi, ssi);
    float xj = zj[(size_t)d * MCOLS]; ssj = fmaf(xj, xj, ssj);
  }
  float rni = 1.0f / fmaxf(sqrtf(ssi), 1e-8f);
  float rnj = 1.0f / fmaxf(sqrtf(ssj), 1e-8f);

  size_t rowI = ((size_t)b * NROWS + m) * DDIM;
  size_t rowJ = ((size_t)b * NROWS + MCOLS + m) * DDIM;
  float dot = 0.f;
  for (int d = 0; d < DDIM; ++d) {
    float xi = zi[(size_t)d * MCOLS] * rni;
    float xj = zj[(size_t)d * MCOLS] * rnj;
    dot = fmaf(xi, xj, dot);
    unsigned short hi_i = f32_to_bf16_rne(xi);
    unsigned short hi_j = f32_to_bf16_rne(xj);
    znHi[rowI + d] = hi_i;
    znHi[rowJ + d] = hi_j;
    znLo[rowI + d] = f32_to_bf16_rne(xi - bf16_to_f32(hi_i));
    znLo[rowJ + d] = f32_to_bf16_rne(xj - bf16_to_f32(hi_j));
  }
  float p = 2.0f * dot;                      // sim / TEMP, TEMP = 0.5
  pos[(size_t)b * NROWS + m]         = p;    // pos is symmetric across pair
  pos[(size_t)b * NROWS + MCOLS + m] = p;
}

// ---------------------------------------------------------------------------
// Phase 2: per 16-row strip, sweep all 256 column tiles. A fragments (full
// K=128, hi+lo) live in registers for the whole strip. 12 WMMAs / tile.
// sim = 2*dot, diag = 2 exactly -> sum exp(sim-2), subtract 1 for diagonal.
// ---------------------------------------------------------------------------
__global__ void __launch_bounds__(128, 1)
simclr_sim_lse(const unsigned short* __restrict__ znHi,
               const unsigned short* __restrict__ znLo,
               const float* __restrict__ pos,
               float* __restrict__ partials) {
  const int lane = threadIdx.x & 31;
  const int wave = threadIdx.x >> 5;
  const int blocksPerSample = NROWS / 64;            // 64 (4 strips per block)
  const int b   = blockIdx.x / blocksPerSample;
  const int blk = blockIdx.x % blocksPerSample;
  const int r0  = blk * 64 + wave * 16;              // strip's first row

  const size_t sOff = (size_t)b * NROWS * DDIM;
  const unsigned short* hiB = znHi + sOff;
  const unsigned short* loB = znLo + sOff;

  // A fragment k order chosen here (contiguous; B uses the same): lanes 0-15 row M=lane, K {c..c+15};
  // lanes 16-31 row M=lane-16, K {c+16..c+31}. c = kc*32.
  const int aRow = r0 + (lane & 15);
  const int aK   = (lane >> 4) * 16;
  v16bf aHi[4], aLo[4];
#pragma unroll
  for (int kc = 0; kc < 4; ++kc) {
    const unsigned short* p = hiB + (size_t)aRow * DDIM + kc * 32 + aK;
    const unsigned short* q = loB + (size_t)aRow * DDIM + kc * 32 + aK;
    FragA fa, fl;
    fa.h[0] = *(const u32x4*)(p);      fa.h[1] = *(const u32x4*)(p + 8);
    fl.h[0] = *(const u32x4*)(q);      fl.h[1] = *(const u32x4*)(q + 8);
    aHi[kc] = fa.b;  aLo[kc] = fl.b;
  }

  // B fragment k order chosen here (contiguous, same as A): lanes 0-15 col N=lane K 0-15,
  // lanes 16-31 col N=lane-16 K 16-31 -> one contiguous 32B load per lane.
  const int bCol = lane & 15;
  const int bK   = (lane >> 4) * 16;

  v8f rs = {};                                       // per-slot exp partial sums
  for (int ct = 0; ct < NROWS / 16; ++ct) {
    const unsigned short* bp = hiB + (size_t)(ct * 16 + bCol) * DDIM + bK;
    const unsigned short* bq = loB + (size_t)(ct * 16 + bCol) * DDIM + bK;
    v8f c = {};
#pragma unroll
    for (int kc = 0; kc < 4; ++kc) {
      FragB bh, bl;
      bh.u = *(const u32x8*)(bp + kc * 32);
      bl.u = *(const u32x8*)(bq + kc * 32);
      // fp32 emulation: hi*hi + lo*hi + hi*lo (lo*lo term negligible)
      c = __builtin_amdgcn_wmma_f32_16x16x32_bf16(false, aHi[kc], false, bh.b,
                                                  (short)0, c, false, false);
      c = __builtin_amdgcn_wmma_f32_16x16x32_bf16(false, aLo[kc], false, bh.b,
                                                  (short)0, c, false, false);
      c = __builtin_amdgcn_wmma_f32_16x16x32_bf16(false, aHi[kc], false, bl.b,
                                                  (short)0, c, false, false);
    }
#pragma unroll
    for (int k = 0; k < 8; ++k)
      rs[k] += __expf(2.0f * c[k] - 2.0f);           // exp(sim - 2), sim <= 2
  }

  // C layout: VGPR k -> row r0+k (lanes 0-15, col=lane) / row r0+8+k
  // (lanes 16-31, col=lane-16). Reduce each 16-lane half.
  float lossAcc = 0.f;
#pragma unroll
  for (int k = 0; k < 8; ++k) {
    float s = rs[k];
    s += __shfl_xor(s, 1, 32);
    s += __shfl_xor(s, 2, 32);
    s += __shfl_xor(s, 4, 32);
    s += __shfl_xor(s, 8, 32);
    if (lane == 0 || lane == 16) {
      int row = r0 + k + ((lane == 16) ? 8 : 0);
      float lse = __logf(fmaxf(s - 1.0f, 1e-30f)) + 2.0f;  // drop diagonal
      lossAcc += lse - pos[(size_t)b * NROWS + row];
    }
  }
  lossAcc += __shfl(lossAcc, 16, 32);                // lane0 += lane16's rows
  if (lane == 0)
    partials[(size_t)b * (NROWS / 16) + blk * 4 + wave] = lossAcc;
}

// ---------------------------------------------------------------------------
// Phase 3: deterministic fixed-order reduction of 4096 strip partials.
// ---------------------------------------------------------------------------
__global__ void simclr_reduce(const float* __restrict__ partials,
                              float* __restrict__ out) {
  __shared__ float sm[256];
  float s = 0.f;
  for (int i = threadIdx.x; i < BATCH * (NROWS / 16); i += 256)
    s += partials[i];
  sm[threadIdx.x] = s;
  __syncthreads();
  for (int off = 128; off > 0; off >>= 1) {
    if ((int)threadIdx.x < off) sm[threadIdx.x] += sm[threadIdx.x + off];
    __syncthreads();
  }
  if (threadIdx.x == 0)
    out[0] = sm[0] / (float)(BATCH * NROWS);   // mean over rows == mean of means
}

extern "C" void kernel_launch(void* const* d_in, const int* in_sizes, int n_in,
                              void* d_out, int out_size, void* d_ws, size_t ws_size,
                              hipStream_t stream) {
  (void)in_sizes; (void)n_in; (void)out_size; (void)ws_size;
  const float* Zi = (const float*)d_in[0];
  const float* Zj = (const float*)d_in[1];
  float* out = (float*)d_out;

  // Workspace carve-up (~33.8 MB total)
  size_t znElems = (size_t)BATCH * NROWS * DDIM;               // 8,388,608
  unsigned short* znHi = (unsigned short*)d_ws;                // 16 MB
  unsigned short* znLo = znHi + znElems;                       // 16 MB
  float* pos      = (float*)(znLo + znElems);                  // 256 KB
  float* partials = pos + (size_t)BATCH * NROWS;               // 16 KB

  simclr_normalize<<<(BATCH * MCOLS + 255) / 256, 256, 0, stream>>>(
      Zi, Zj, znHi, znLo, pos);
  simclr_sim_lse<<<BATCH * (NROWS / 64), 128, 0, stream>>>(
      znHi, znLo, pos, partials);
  simclr_reduce<<<1, 256, 0, stream>>>(partials, out);
}
